// Block_27127013442041
// MI455X (gfx1250) — compile-verified, matrix operand order checked by reading, not run on hardware
//
#include <hip/hip_runtime.h>

typedef __attribute__((ext_vector_type(16))) _Float16 v16h;
typedef __attribute__((ext_vector_type(8)))  _Float16 v8h;
typedef __attribute__((ext_vector_type(8)))  float    v8f;

#define NN    8192
#define EE    131072
#define NRESC 1024
#define LEAKC 0.1f
#define R0C   5.0f
#define PIC   3.14159265358979323846f

// ---------------- f16 weight slots (each 4096 halves inside ws, fragment-ordered) --------
enum {
  S_DT0_W1 = 0, S_DT0_W2, S_DT0_W3, S_DT0_WV,
  S_DT1_W1, S_DT1_W2, S_DT1_W3, S_DT1_WV,
  S_STP_Y000, S_STP_Y110, S_STP_Y011, S_STP_Y101, S_STP_Y111,
  S_STP_L0, S_STP_R0, S_STP_L1, S_STP_R1, S_STP_UL, S_STP_UR,
  S_STP_W1, S_STP_W2, S_STP_W3,
  S_M0_Y000, S_M0_Y110, S_M0_Y011, S_M0_Y101, S_M0_Y111, S_M0_ENCW, S_M0_W1, S_M0_W2, S_M0_W3,
  S_M1_Y000, S_M1_Y110, S_M1_Y011, S_M1_Y101, S_M1_Y111, S_M1_ENCW, S_M1_W1, S_M1_W2, S_M1_W3,
  NSLOT
};

struct W16Args { const float* src[NSLOT]; int ksz[NSLOT]; int nout[NSLOT]; };

__device__ __forceinline__ float sigf(float x) { return 1.0f / (1.0f + expf(-x)); }

// A fragment from LDS: two contiguous 8-half runs per lane (ISA 7.12.2 16-bit A layout).
// Requires A rows 16B-aligned (lda in halves must be multiple of 8; base __align__(16)).
__device__ __forceinline__ v16h fragA(const _Float16* A, int lda, int kt, int lane) {
  int m = lane & 15, hi = (lane >> 4) & 1;
  const _Float16* p = A + m * lda + kt * 32 + hi * 8;
  v8h lo = *(const v8h*)(p);
  v8h hh = *(const v8h*)(p + 16);
  return __builtin_shufflevector(lo, hh, 0,1,2,3,4,5,6,7,8,9,10,11,12,13,14,15);
}

// ---------------- wave-level WMMA matmul: C(16xNOUT) = A(16xK) * B(KxNOUT) ----------------
// A: LDS f16 row-major (leading dim lda, halves). B: global f16 in FRAGMENT order:
// frag fi = kt*(NOUT/16)+nt occupies halves [fi*512, fi*512+512), lane-major 16 halves each.
template<int K, int NOUT>
__device__ __forceinline__ void wave_mm(const _Float16* A, int lda,
                                        const _Float16* __restrict__ B,
                                        float* C, int ldc, int lane) {
  const int n  = lane & 15;
  const int hi = (lane >> 4) & 1;
  const v16h* Bf = (const v16h*)B;
#pragma unroll
  for (int nt = 0; nt < NOUT / 16; ++nt) {
    v8f acc = {};
#pragma unroll
    for (int kt = 0; kt < K / 32; ++kt) {
      v16h a = fragA(A, lda, kt, lane);
      v16h b = Bf[(kt * (NOUT / 16) + nt) * 32 + lane];
      acc = __builtin_amdgcn_wmma_f32_16x16x32_f16(false, a, false, b,
                                                   (short)0, acc, false, false);
    }
#pragma unroll
    for (int r = 0; r < 8; ++r)
      C[(hi * 8 + r) * ldc + nt * 16 + n] = acc[r];   // D: row = 8*hi + r, col = lane&15
  }
}

// mlp: h=lrelu(x@W1+b1); h=lrelu(h@W2+b2); out=h@W3.  Input f32 tile `in` (16x64),
// staging in sA (lda=72), result left in sC (16x64).
__device__ __forceinline__ void mlp3(const float* in, _Float16* sA, float* sC,
                                     const _Float16* W1, const _Float16* W2, const _Float16* W3,
                                     const float* b1, const float* b2, int lane) {
  for (int i = lane; i < 1024; i += 32) { int e=i>>6, d=i&63; sA[e*72+d] = (_Float16)in[e*64+d]; }
  __syncthreads();
  wave_mm<64,64>(sA, 72, W1, sC, 64, lane);
  __syncthreads();
  for (int i = lane; i < 1024; i += 32) { int e=i>>6, d=i&63;
    float v = sC[e*64+d] + b1[d]; v = v > 0.f ? v : LEAKC*v; sA[e*72+d] = (_Float16)v; }
  __syncthreads();
  wave_mm<64,64>(sA, 72, W2, sC, 64, lane);
  __syncthreads();
  for (int i = lane; i < 1024; i += 32) { int e=i>>6, d=i&63;
    float v = sC[e*64+d] + b2[d]; v = v > 0.f ? v : LEAKC*v; sA[e*72+d] = (_Float16)v; }
  __syncthreads();
  wave_mm<64,64>(sA, 72, W3, sC, 64, lane);
  __syncthreads();
}

// ---------------- tiny utility kernels ----------------
__global__ void k_zero(float* p, int n) {
  for (int i = blockIdx.x * blockDim.x + threadIdx.x; i < n; i += gridDim.x * blockDim.x) p[i] = 0.f;
}
__global__ void k_copy(float* dst, const float* src, int n) {
  for (int i = blockIdx.x * blockDim.x + threadIdx.x; i < n; i += gridDim.x * blockDim.x) dst[i] = src[i];
}
// Convert f32 weight (K x N, row-major in x out) into f16 WMMA-fragment order.
__global__ void k_cvt(W16Args a, _Float16* hw) {
  int slot = blockIdx.x;
  const float* s = a.src[slot];
  int N = a.nout[slot], K = a.ksz[slot];
  _Float16* d = hw + (size_t)slot * 4096;
  int ntiles = N >> 4;
  int nfr = (K >> 5) * ntiles;
  int lane = threadIdx.x & 31;
  int n = lane & 15, hi = (lane >> 4) & 1;
  for (int fi = (int)(threadIdx.x >> 5); fi < nfr; fi += (int)(blockDim.x >> 5)) {
    int kt = fi / ntiles, nt = fi % ntiles;
    for (int h = 0; h < 16; ++h) {
      int k = kt * 32 + ((h >> 3) << 4) + hi * 8 + (h & 7);
      d[fi * 512 + lane * 16 + h] = (_Float16)s[k * N + nt * 16 + n];
    }
  }
}
__global__ void k_res_emb(const int* __restrict__ at, const float* __restrict__ tab, float* re) {
  int i = blockIdx.x * blockDim.x + threadIdx.x;
  if (i < NN * 64) { int n = i >> 6, d = i & 63; re[i] = tab[at[n] * 64 + d]; }
}
__global__ void k_time_embed(const float* t, const float* W, const float* b, float* te) {
  int o = threadIdx.x; float acc = 0.f; float tv = t[0];
  for (int j = 0; j < 32; ++j) {
    float ang = expf(-(float)j / 31.0f); float ph = tv * ang;
    acc += sinf(ph) * W[j * 64 + o] + cosf(ph) * W[(32 + j) * 64 + o];
  }
  te[o] = acc + b[o];
}
__global__ void k_center_acc(const float* p0, const float* p1, const int* ridx,
                             float* cnt, float* c0, float* c1) {
  int n = blockIdx.x * blockDim.x + threadIdx.x;
  if (n >= NN) return;
  int r = ridx[n];
  atomicAdd(&cnt[r], 1.0f);
  for (int c = 0; c < 3; ++c) {
    atomicAdd(&c0[r * 3 + c], p0[n * 3 + c]);
    atomicAdd(&c1[r * 3 + c], p1[n * 3 + c]);
  }
}
__global__ void k_center_fin(const float* cnt, float* c0, float* c1) {
  int r = blockIdx.x * blockDim.x + threadIdx.x;
  if (r >= NRESC) return;
  float inv = 1.0f / fmaxf(cnt[r], 1.0f);
  for (int c = 0; c < 3; ++c) { c0[r * 3 + c] *= inv; c1[r * 3 + c] *= inv; }
}

// ---------------- disp_tensors: per-edge MLP + scatter (WMMA) ----------------
__global__ __launch_bounds__(32) void k_disp(
    const float* __restrict__ pos, const int* __restrict__ src, const int* __restrict__ dst,
    const float* __restrict__ box, const float* __restrict__ re,
    const _Float16* __restrict__ hw, int wslot,
    const float* __restrict__ b1, const float* __restrict__ b2,
    float* xa, float* xv) {
  __shared__ __align__(16) _Float16 sA[16 * 72];
  __shared__ float sC[16 * 64];
  __shared__ float sRad[16 * 64];
  __shared__ float sRS[16][4];
  __shared__ float sDist[16];
  __shared__ int sSrc[16], sDst[16];
  const _Float16* W1 = hw + (size_t)(wslot + 0) * 4096;
  const _Float16* W2 = hw + (size_t)(wslot + 1) * 4096;
  const _Float16* W3 = hw + (size_t)(wslot + 2) * 4096;
  const _Float16* Wv = hw + (size_t)(wslot + 3) * 4096;
  int lane = threadIdx.x;
  int e0 = blockIdx.x * 16;
  if (lane < 16) {
    int e = e0 + lane; int s = src[e], d = dst[e];
    sSrc[lane] = s; sDst[lane] = d;
    float r[3], nr2 = 0.f;
    for (int c = 0; c < 3; ++c) {
      float bx = box[c];
      float v = pos[d * 3 + c] - pos[s * 3 + c];
      v = v - bx * rintf(v / bx);
      r[c] = v;
    }
    sDist[lane] = sqrtf(r[0]*r[0] + r[1]*r[1] + r[2]*r[2]);
    float v3[3];
    for (int c = 0; c < 3; ++c) { v3[c] = r[c] * (7.0f / R0C); nr2 += v3[c]*v3[c]; }
    float nr = sqrtf(nr2), sc = tanhf(nr) / fmaxf(nr, 1e-6f);
    for (int c = 0; c < 3; ++c) sRS[lane][c] = v3[c] * sc;
  }
  __syncthreads();
  { // radial_encode + res_emb gather -> f16 A tile
    int e = lane & 15, hi = lane >> 4;
    int de = sDst[e]; float dist = sDist[e];
    for (int j = 0; j < 32; ++j) {
      float ph = (PIC / R0C) * (float)(1 + (j >> 1)) * dist;
      float v = hi ? sinf(ph) : cosf(ph);
      int dim = hi * 32 + j;
      sA[e * 72 + dim] = (_Float16)(v + re[de * 64 + dim]);
    }
  }
  __syncthreads();
  wave_mm<64,64>(sA, 72, W1, sC, 64, lane);
  __syncthreads();
  for (int i = lane; i < 1024; i += 32) { int e=i>>6, d=i&63;
    float v = sC[e*64+d] + b1[d]; v = v > 0.f ? v : LEAKC*v; sA[e*72+d] = (_Float16)v; }
  __syncthreads();
  wave_mm<64,64>(sA, 72, W2, sC, 64, lane);
  __syncthreads();
  for (int i = lane; i < 1024; i += 32) { int e=i>>6, d=i&63;
    float v = sC[e*64+d] + b2[d]; v = v > 0.f ? v : LEAKC*v; sA[e*72+d] = (_Float16)v; }
  __syncthreads();
  wave_mm<64,64>(sA, 72, W3, sC, 64, lane);
  __syncthreads();
  for (int i = lane; i < 1024; i += 32) { int e=i>>6, d=i&63;
    float v = sC[e*64+d]; sRad[e*64+d] = v; sA[e*72+d] = (_Float16)v; }
  __syncthreads();
  wave_mm<64,32>(sA, 72, Wv, sC, 64, lane);   // radWv (16x32)
  __syncthreads();
  for (int i = lane; i < 1024; i += 32) { int e=i>>6, d=i&63;
    atomicAdd(&xa[sSrc[e] * 64 + d], sRad[e*64+d]); }
  { int e = lane & 15, hi = lane >> 4; int s = sSrc[e];
    for (int o = hi * 16; o < hi * 16 + 16; ++o) {
      float rv = sC[e*64+o];
      for (int c = 0; c < 3; ++c) atomicAdd(&xv[s * 96 + o * 3 + c], rv * sRS[e][c]);
    }
  }
}

// ---------------- self_tens_prods (per 16-node tile, WMMA) ----------------
__global__ __launch_bounds__(32) void k_stp(
    float* xa, float* xv, const float* __restrict__ re, const float* __restrict__ te,
    const float* __restrict__ p0, const float* __restrict__ p1,
    const float* __restrict__ c0, const float* __restrict__ c1, const int* __restrict__ ridx,
    const float* __restrict__ pW, const float* __restrict__ r0W, const float* __restrict__ r1W,
    const _Float16* __restrict__ hw,
    const float* __restrict__ b1, const float* __restrict__ b2,
    const float* __restrict__ gag, const float* __restrict__ gab, const float* __restrict__ gvg) {
  __shared__ __align__(16) _Float16 sA[16 * 72];
  __shared__ float sC[16 * 64];
  __shared__ float sL0[16 * 64], sR0[16 * 64];
  __shared__ float sL1[16 * 192], sR1[16 * 192];
  __shared__ float sYa[16 * 64];
  __shared__ float sYv[16 * 96];
  const _Float16* L0 = hw + (size_t)S_STP_L0 * 4096;
  const _Float16* R0 = hw + (size_t)S_STP_R0 * 4096;
  const _Float16* L1 = hw + (size_t)S_STP_L1 * 4096;
  const _Float16* R1 = hw + (size_t)S_STP_R1 * 4096;
  const _Float16* Y000 = hw + (size_t)S_STP_Y000 * 4096;
  const _Float16* Y110 = hw + (size_t)S_STP_Y110 * 4096;
  const _Float16* Y011 = hw + (size_t)S_STP_Y011 * 4096;
  const _Float16* Y101 = hw + (size_t)S_STP_Y101 * 4096;
  const _Float16* Y111 = hw + (size_t)S_STP_Y111 * 4096;
  const _Float16* UL = hw + (size_t)S_STP_UL * 4096;
  const _Float16* UR = hw + (size_t)S_STP_UR * 4096;
  const _Float16* W1 = hw + (size_t)S_STP_W1 * 4096;
  const _Float16* W2 = hw + (size_t)S_STP_W2 * 4096;
  const _Float16* W3 = hw + (size_t)S_STP_W3 * 4096;
  int lane = threadIdx.x;
  int n0 = blockIdx.x * 16;
  // sa = xa + res_emb + te
  for (int i = lane; i < 1024; i += 32) { int e=i>>6, d=i&63; int n=n0+e;
    sA[e*72+d] = (_Float16)(xa[n*64+d] + re[n*64+d] + te[d]); }
  __syncthreads();
  wave_mm<64,64>(sA, 72, L0, sC, 64, lane); __syncthreads();
  for (int i = lane; i < 1024; i += 32) sL0[i] = sC[i];
  __syncthreads();
  wave_mm<64,64>(sA, 72, R0, sC, 64, lane); __syncthreads();
  for (int i = lane; i < 1024; i += 32) sR0[i] = sC[i];
  __syncthreads();
  // sv = xv + pe + rp, then l1/r1 per component
  for (int c = 0; c < 3; ++c) {
    for (int i = lane; i < 512; i += 32) { int e=i>>5, o=i&31; int n=n0+e; int r=ridx[n];
      float dp = 0.1f * (p1[n*3+c] - p0[n*3+c]);
      float rel0 = p0[n*3+c] - c0[r*3+c];
      float rel1 = p1[n*3+c] - c1[r*3+c];
      float v = xv[n*96 + o*3 + c] + dp * pW[o] + 0.2f * (rel0 * r0W[o] + rel1 * r1W[o]);
      sA[e*72+o] = (_Float16)v; }
    __syncthreads();
    wave_mm<32,64>(sA, 72, L1, sC, 64, lane); __syncthreads();
    for (int i = lane; i < 1024; i += 32) { int e=i>>6, d=i&63; sL1[e*192 + d*3 + c] = sC[e*64+d]; }
    __syncthreads();
    wave_mm<32,64>(sA, 72, R1, sC, 64, lane); __syncthreads();
    for (int i = lane; i < 1024; i += 32) { int e=i>>6, d=i&63; sR1[e*192 + d*3 + c] = sC[e*64+d]; }
    __syncthreads();
  }
  // y_a = (l0*r0)@Y000 + (sum_c l1*r1)@Y110
  for (int i = lane; i < 1024; i += 32) { int e=i>>6, d=i&63;
    sA[e*72+d] = (_Float16)(sL0[i] * sR0[i]); }
  __syncthreads();
  wave_mm<64,64>(sA, 72, Y000, sC, 64, lane); __syncthreads();
  for (int i = lane; i < 1024; i += 32) sYa[i] = sC[i];
  __syncthreads();
  for (int i = lane; i < 1024; i += 32) { int e=i>>6, d=i&63;
    float s = 0.f;
    for (int c = 0; c < 3; ++c) s += sL1[e*192+d*3+c] * sR1[e*192+d*3+c];
    sA[e*72+d] = (_Float16)s; }
  __syncthreads();
  wave_mm<64,64>(sA, 72, Y110, sC, 64, lane); __syncthreads();
  for (int i = lane; i < 1024; i += 32) sYa[i] += sC[i];
  __syncthreads();
  // y_a += mlp(y_a)
  mlp3(sYa, sA, sC, W1, W2, W3, b1, b2, lane);
  for (int i = lane; i < 1024; i += 32) sYa[i] += sC[i];
  __syncthreads();
  // l0 += y_a@UL ; r0 += y_a@UR
  for (int i = lane; i < 1024; i += 32) { int e=i>>6, d=i&63; sA[e*72+d] = (_Float16)sYa[i]; }
  __syncthreads();
  wave_mm<64,64>(sA, 72, UL, sC, 64, lane); __syncthreads();
  for (int i = lane; i < 1024; i += 32) sL0[i] += sC[i];
  __syncthreads();
  wave_mm<64,64>(sA, 72, UR, sC, 64, lane); __syncthreads();
  for (int i = lane; i < 1024; i += 32) sR0[i] += sC[i];
  __syncthreads();
  // y_v per component
  for (int c = 0; c < 3; ++c) {
    int ca = (c + 1) % 3, cb = (c + 2) % 3;
    for (int i = lane; i < 1024; i += 32) { int e=i>>6, d=i&63;
      sA[e*72+d] = (_Float16)(sL0[i] * sR1[e*192+d*3+c]); }
    __syncthreads();
    wave_mm<64,32>(sA, 72, Y011, sC, 64, lane); __syncthreads();
    for (int i = lane; i < 512; i += 32) { int e=i>>5, o=i&31; sYv[e*96+o*3+c] = sC[e*64+o]; }
    __syncthreads();
    for (int i = lane; i < 1024; i += 32) { int e=i>>6, d=i&63;
      sA[e*72+d] = (_Float16)(sL1[e*192+d*3+c] * sR0[i]); }
    __syncthreads();
    wave_mm<64,32>(sA, 72, Y101, sC, 64, lane); __syncthreads();
    for (int i = lane; i < 512; i += 32) { int e=i>>5, o=i&31; sYv[e*96+o*3+c] += sC[e*64+o]; }
    __syncthreads();
    for (int i = lane; i < 1024; i += 32) { int e=i>>6, d=i&63;
      float q = sL1[e*192+d*3+ca] * sR1[e*192+d*3+cb] - sL1[e*192+d*3+cb] * sR1[e*192+d*3+ca];
      sA[e*72+d] = (_Float16)q; }
    __syncthreads();
    wave_mm<64,32>(sA, 72, Y111, sC, 64, lane); __syncthreads();
    for (int i = lane; i < 512; i += 32) { int e=i>>5, o=i&31; sYv[e*96+o*3+c] += sC[e*64+o]; }
    __syncthreads();
  }
  // gn0 on y_a (8 groups of 8) -> xa +=
  for (int g = lane; g < 128; g += 32) {
    int e = g >> 3, grp = g & 7; int n = n0 + e;
    float m = 0.f, m2 = 0.f;
    for (int j = 0; j < 8; ++j) { float v = sYa[e*64 + grp*8 + j]; m += v; m2 += v*v; }
    m *= 0.125f; m2 *= 0.125f;
    float inv = rsqrtf(m2 - m*m + 1e-5f);
    for (int j = 0; j < 8; ++j) {
      int d = grp*8 + j;
      xa[n*64 + d] += (sYa[e*64+d] - m) * inv * gag[d] + gab[d];
    }
  }
  // gn1 on y_v (8 groups of 4 dims x 3 comps) -> xv +=
  for (int g = lane; g < 128; g += 32) {
    int e = g >> 3, grp = g & 7; int n = n0 + e;
    float ms = 0.f;
    for (int j = 0; j < 4; ++j)
      for (int c = 0; c < 3; ++c) { float v = sYv[e*96 + (grp*4+j)*3 + c]; ms += v*v; }
    float inv = rsqrtf(ms / 12.0f + 1e-5f);
    for (int j = 0; j < 4; ++j)
      for (int c = 0; c < 3; ++c) {
        int o = grp*4 + j;
        xv[n*96 + o*3 + c] += sYv[e*96 + o*3 + c] * inv * gvg[o];
      }
  }
}

// ---------------- aminos_conv helpers (scalar; small FLOPs) ----------------
__global__ __launch_bounds__(64) void k_ac_gather(
    const float* __restrict__ xa, const float* __restrict__ xv, const float* __restrict__ re,
    const float* __restrict__ riaG, const float* __restrict__ riaW,
    const float* __restrict__ rivG, const float* __restrict__ rivW,
    const int* __restrict__ ridx, float* yaR, float* yvR) {
  __shared__ float sre[64];
  __shared__ float sxg[64];
  int n = blockIdx.x, t = threadIdx.x;
  sre[t] = re[n * 64 + t];
  __syncthreads();
  float g = 0.f;
  for (int j = 0; j < 64; ++j) g += sre[j] * riaG[j * 64 + t];
  sxg[t] = xa[n * 64 + t] * sigf(g);
  __syncthreads();
  int r = ridx[n];
  float acc = 0.f;
  for (int i = 0; i < 64; ++i) acc += sxg[i] * riaW[i * 64 + t];
  atomicAdd(&yaR[r * 64 + t], acc);
  if (t < 32) {
    float gv = 0.f;
    for (int j = 0; j < 64; ++j) gv += sre[j] * rivG[j * 32 + t];
    gv = sigf(gv);
    for (int c = 0; c < 3; ++c) {
      float tv = 0.f;
      for (int i = 0; i < 32; ++i) tv += rivW[i * 32 + t] * xv[n * 96 + i * 3 + c];
      atomicAdd(&yvR[r * 96 + t * 3 + c], tv * gv);
    }
  }
}
__global__ __launch_bounds__(64) void k_conv_a(const float* __restrict__ in,
                                               const float* __restrict__ w,
                                               float* out, int lrelu) {
  int r = blockIdx.x, o = threadIdx.x;
  float acc = 0.f;
  for (int dk = 0; dk < 3; ++dk) {
    int rr = r + dk - 1;
    if (rr < 0 || rr >= NRESC) continue;
    const float* xp = &in[rr * 64];
    for (int i = 0; i < 64; ++i) acc += w[o * 192 + i * 3 + dk] * xp[i];
  }
  if (lrelu) acc = acc > 0.f ? acc : LEAKC * acc;
  out[r * 64 + o] = acc;
}
__global__ __launch_bounds__(96) void k_conv_v(const float* __restrict__ in,
                                               const float* __restrict__ w, float* out) {
  int r = blockIdx.x, t = threadIdx.x, o = t / 3, c = t % 3;
  float acc = 0.f;
  for (int dk = 0; dk < 5; ++dk) {
    int rr = r + dk - 2;
    if (rr < 0 || rr >= NRESC) continue;
    for (int i = 0; i < 32; ++i) acc += w[o * 160 + i * 5 + dk] * in[rr * 96 + i * 3 + c];
  }
  out[r * 96 + o * 3 + c] = acc;
}
__global__ __launch_bounds__(64) void k_ac_out(
    float* xa, float* xv, const float* __restrict__ yaC, const float* __restrict__ yvC,
    const float* __restrict__ re,
    const float* __restrict__ roaW, const float* __restrict__ roaG,
    const float* __restrict__ rovW, const float* __restrict__ rovG,
    const int* __restrict__ ridx) {
  __shared__ float sre[64];
  int n = blockIdx.x, t = threadIdx.x;
  sre[t] = re[n * 64 + t];
  __syncthreads();
  int r = ridx[n];
  float a = 0.f;
  for (int i = 0; i < 64; ++i) a += yaC[r * 64 + i] * roaW[i * 64 + t];
  float g = 0.f;
  for (int j = 0; j < 64; ++j) g += sre[j] * roaG[j * 64 + t];
  xa[n * 64 + t] += 0.2f * a * sigf(g);
  if (t < 32) {
    float gv = 0.f;
    for (int j = 0; j < 64; ++j) gv += sre[j] * rovG[j * 32 + t];
    gv = sigf(gv);
    for (int c = 0; c < 3; ++c) {
      float av = 0.f;
      for (int i = 0; i < 32; ++i) av += rovW[i * 32 + t] * yvC[r * 96 + i * 3 + c];
      xv[n * 96 + t * 3 + c] += 0.2f * av * gv;
    }
  }
}

// ---------------- messages: per-node precompute la/lv, then per-edge WMMA ----------------
__global__ __launch_bounds__(64) void k_msg_pre(const float* __restrict__ xa,
                                                const float* __restrict__ xv,
                                                const float* __restrict__ L0,
                                                const float* __restrict__ L1,
                                                float* la, float* lv) {
  int n = blockIdx.x, t = threadIdx.x;
  float a = 0.f;
  for (int i = 0; i < 64; ++i) a += xa[n * 64 + i] * L0[i * 64 + t];
  la[n * 64 + t] = a;
  for (int c = 0; c < 3; ++c) {
    float v = 0.f;
    for (int i = 0; i < 32; ++i) v += L1[i * 64 + t] * xv[n * 96 + i * 3 + c];
    lv[n * 192 + t * 3 + c] = v;
  }
}

__global__ __launch_bounds__(32) void k_msg(
    const float* __restrict__ pos, const int* __restrict__ src, const int* __restrict__ dst,
    const float* __restrict__ box, const float* __restrict__ la, const float* __restrict__ lv,
    const _Float16* __restrict__ hw, int wslot,
    const float* __restrict__ encb, const float* __restrict__ b1, const float* __restrict__ b2,
    float* out_a, float* out_v) {
  __shared__ __align__(16) _Float16 sA[16 * 72];
  __shared__ float sC[16 * 64];
  __shared__ float sRad[16 * 64];
  __shared__ float sLa[16 * 64];
  __shared__ float sPA[16 * 64];
  __shared__ float sPV[16 * 96];
  __shared__ float sRS[16][4];
  __shared__ float sDist[16];
  __shared__ int sSrc[16], sDst[16];
  const _Float16* Y000 = hw + (size_t)(wslot + 0) * 4096;
  const _Float16* Y110 = hw + (size_t)(wslot + 1) * 4096;
  const _Float16* Y011 = hw + (size_t)(wslot + 2) * 4096;
  const _Float16* Y101 = hw + (size_t)(wslot + 3) * 4096;
  const _Float16* Y111 = hw + (size_t)(wslot + 4) * 4096;
  const _Float16* ENCW = hw + (size_t)(wslot + 5) * 4096;
  const _Float16* W1   = hw + (size_t)(wslot + 6) * 4096;
  const _Float16* W2   = hw + (size_t)(wslot + 7) * 4096;
  const _Float16* W3   = hw + (size_t)(wslot + 8) * 4096;
  int lane = threadIdx.x;
  int e0 = blockIdx.x * 16;
  if (lane < 16) {
    int e = e0 + lane; int s = src[e], d = dst[e];
    sSrc[lane] = s; sDst[lane] = d;
    float r[3], nr2 = 0.f;
    for (int c = 0; c < 3; ++c) {
      float bx = box[c];
      float v = pos[d * 3 + c] - pos[s * 3 + c];
      v = v - bx * rintf(v / bx);
      r[c] = v;
    }
    sDist[lane] = sqrtf(r[0]*r[0] + r[1]*r[1] + r[2]*r[2]);
    float v3[3];
    for (int c = 0; c < 3; ++c) { v3[c] = r[c] * (7.0f / R0C); nr2 += v3[c]*v3[c]; }
    float nr = sqrtf(nr2), sc = tanhf(nr) / fmaxf(nr, 1e-6f);
    for (int c = 0; c < 3; ++c) sRS[lane][c] = v3[c] * sc;
  }
  __syncthreads();
  { int e = lane & 15, hi = lane >> 4; float dist = sDist[e];
    for (int j = 0; j < 32; ++j) {
      float ph = (PIC / R0C) * (float)(1 + (j >> 1)) * dist;
      sA[e * 72 + hi * 32 + j] = (_Float16)(hi ? sinf(ph) : cosf(ph));
    }
  }
  __syncthreads();
  wave_mm<64,64>(sA, 72, ENCW, sC, 64, lane);   // rad = enc(radial) + enc_b
  __syncthreads();
  for (int i = lane; i < 1024; i += 32) { int e=i>>6, d=i&63; sRad[i] = sC[i] + encb[d]; }
  __syncthreads();
  // A = la*rad -> psi_a term1 (Y000) and t011 (Y011)
  for (int i = lane; i < 1024; i += 32) { int e=i>>6, d=i&63;
    float lav = la[sDst[e] * 64 + d];
    sLa[i] = lav;
    sA[e*72+d] = (_Float16)(lav * sRad[i]); }
  __syncthreads();
  wave_mm<64,64>(sA, 72, Y000, sC, 64, lane); __syncthreads();
  for (int i = lane; i < 1024; i += 32) sPA[i] = sC[i];
  __syncthreads();
  wave_mm<64,32>(sA, 72, Y011, sC, 64, lane); __syncthreads();
  for (int i = lane; i < 512; i += 32) { int e=i>>5, o=i&31;
    float t011 = sC[e*64+o];
    for (int c = 0; c < 3; ++c) sPV[e*96+o*3+c] = t011 * sRS[e][c]; }
  __syncthreads();
  // A = rad * (lv . rs) -> psi_a term2 (Y110)
  for (int i = lane; i < 1024; i += 32) { int e=i>>6, d=i&63;
    const float* lvp = &lv[sDst[e] * 192 + d * 3];
    float dv = lvp[0]*sRS[e][0] + lvp[1]*sRS[e][1] + lvp[2]*sRS[e][2];
    sA[e*72+d] = (_Float16)(sRad[i] * dv); }
  __syncthreads();
  wave_mm<64,64>(sA, 72, Y110, sC, 64, lane); __syncthreads();
  for (int i = lane; i < 1024; i += 32) sPA[i] += sC[i];
  __syncthreads();
  // per-component: Y101 (lv_c*rad) and Y111 (rad*cross(lv, rs)_c)
  for (int c = 0; c < 3; ++c) {
    int ca = (c + 1) % 3, cb = (c + 2) % 3;
    for (int i = lane; i < 1024; i += 32) { int e=i>>6, d=i&63;
      sA[e*72+d] = (_Float16)(lv[sDst[e]*192 + d*3 + c] * sRad[i]); }
    __syncthreads();
    wave_mm<64,32>(sA, 72, Y101, sC, 64, lane); __syncthreads();
    for (int i = lane; i < 512; i += 32) { int e=i>>5, o=i&31; sPV[e*96+o*3+c] += sC[e*64+o]; }
    __syncthreads();
    for (int i = lane; i < 1024; i += 32) { int e=i>>6, d=i&63;
      const float* lvp = &lv[sDst[e]*192 + d*3];
      float q = lvp[ca] * sRS[e][cb] - lvp[cb] * sRS[e][ca];
      sA[e*72+d] = (_Float16)(sRad[i] * q); }
    __syncthreads();
    wave_mm<64,32>(sA, 72, Y111, sC, 64, lane); __syncthreads();
    for (int i = lane; i < 512; i += 32) { int e=i>>5, o=i&31; sPV[e*96+o*3+c] += sC[e*64+o]; }
    __syncthreads();
  }
  // psi_a += mlp(psi_a)
  mlp3(sPA, sA, sC, W1, W2, W3, b1, b2, lane);
  for (int i = lane; i < 1024; i += 32) sPA[i] += sC[i];
  __syncthreads();
  // scatter 0.1 * psi into outputs
  for (int i = lane; i < 1024; i += 32) { int e=i>>6, d=i&63;
    atomicAdd(&out_a[sSrc[e] * 64 + d], 0.1f * sPA[i]); }
  for (int i = lane; i < 1536; i += 32) { int e = i / 96, j = i % 96;
    atomicAdd(&out_v[sSrc[e] * 96 + j], 0.1f * sPV[e*96+j]); }
}

// ---------------- host orchestration ----------------
extern "C" void kernel_launch(void* const* d_in, const int* in_sizes, int n_in,
                              void* d_out, int out_size, void* d_ws, size_t ws_size,
                              hipStream_t stream) {
  (void)in_sizes; (void)n_in; (void)out_size; (void)ws_size;
  auto F  = [&](int i) { return (const float*)d_in[i]; };
  auto Ii = [&](int i) { return (const int*)d_in[i]; };

  char* ws = (char*)d_ws;
  size_t off = 0;
  auto alloc = [&](size_t bytes) { size_t r = off; off += (bytes + 255) & ~((size_t)255); return r; };
  _Float16* hw = (_Float16*)(ws + alloc((size_t)NSLOT * 4096 * 2));
  float* xa  = (float*)(ws + alloc((size_t)NN * 64 * 4));
  float* xv  = (float*)(ws + alloc((size_t)NN * 96 * 4));
  float* re  = (float*)(ws + alloc((size_t)NN * 64 * 4));
  float* te  = (float*)(ws + alloc(64 * 4));
  float* cnt = (float*)(ws + alloc((size_t)NRESC * 4));
  float* c0  = (float*)(ws + alloc((size_t)NRESC * 3 * 4));
  float* c1  = (float*)(ws + alloc((size_t)NRESC * 3 * 4));
  float* yaR = (float*)(ws + alloc((size_t)NRESC * 64 * 4));
  float* hR  = (float*)(ws + alloc((size_t)NRESC * 64 * 4));
  float* yaC = (float*)(ws + alloc((size_t)NRESC * 64 * 4));
  float* yvR = (float*)(ws + alloc((size_t)NRESC * 96 * 4));
  float* yvC = (float*)(ws + alloc((size_t)NRESC * 96 * 4));
  float* la  = (float*)(ws + alloc((size_t)NN * 64 * 4));
  float* lv  = (float*)(ws + alloc((size_t)NN * 192 * 4));

  // f16 weight conversion table (leaf index -> slot, recursive insertion-order flatten)
  W16Args wa{}; int s = 0;
  auto put = [&](int leaf, int K, int N) { wa.src[s] = F(leaf); wa.ksz[s] = K; wa.nout[s] = N; ++s; };
  put(6,64,64);  put(8,64,64);  put(10,64,64); put(11,64,32);          // dt0 W1,W2,W3,Wv
  put(12,64,64); put(14,64,64); put(16,64,64); put(17,64,32);          // dt1
  put(18,64,64); put(19,64,64); put(20,64,32); put(21,64,32); put(22,64,32); // stp Y000..Y111
  put(23,64,64); put(24,64,64); put(25,32,64); put(26,32,64);          // L0,R0,L1,R1
  put(27,64,64); put(28,64,64);                                        // UL,UR
  put(29,64,64); put(31,64,64); put(33,64,64);                         // stp mlp W1,W2,W3
  put(48,64,64); put(49,64,64); put(50,64,32); put(51,64,32); put(52,64,32); // msg0 Y*
  put(55,64,64); put(57,64,64); put(59,64,64); put(61,64,64);          // msg0 encW, mlp
  put(62,64,64); put(63,64,64); put(64,64,32); put(65,64,32); put(66,64,32); // msg1 Y*
  put(69,64,64); put(71,64,64); put(73,64,64); put(75,64,64);          // msg1 encW, mlp
  k_cvt<<<NSLOT, 256, 0, stream>>>(wa, hw);

  k_copy<<<2048, 256, 0, stream>>>(xa, F(79), NN * 64);
  k_copy<<<3072, 256, 0, stream>>>(xv, F(80), NN * 96);
  k_zero<<<28, 256, 0, stream>>>(cnt, NRESC * 7);   // cnt + c0 + c1 contiguous
  k_res_emb<<<2048, 256, 0, stream>>>(Ii(86), F(3), re);
  k_time_embed<<<1, 64, 0, stream>>>(F(76), F(0), F(1), te);
  k_center_acc<<<NN / 256, 256, 0, stream>>>(F(77), F(78), Ii(87), cnt, c0, c1);
  k_center_fin<<<NRESC / 256, 256, 0, stream>>>(cnt, c0, c1);

  k_disp<<<EE / 16, 32, 0, stream>>>(F(77), Ii(82), Ii(83), F(81), re, hw, S_DT0_W1, F(7), F(9), xa, xv);
  k_disp<<<EE / 16, 32, 0, stream>>>(F(78), Ii(84), Ii(85), F(81), re, hw, S_DT1_W1, F(13), F(15), xa, xv);

  k_stp<<<NN / 16, 32, 0, stream>>>(xa, xv, re, te, F(77), F(78), c0, c1, Ii(87),
                                    F(2), F(4), F(5), hw, F(30), F(32), F(34), F(35), F(36));

  k_zero<<<256, 256, 0, stream>>>(yaR, NRESC * 64);
  k_zero<<<384, 256, 0, stream>>>(yvR, NRESC * 96);
  k_ac_gather<<<NN, 64, 0, stream>>>(xa, xv, re, F(37), F(38), F(39), F(40), Ii(87), yaR, yvR);
  k_conv_a<<<NRESC, 64, 0, stream>>>(yaR, F(41), hR, 1);
  k_conv_a<<<NRESC, 64, 0, stream>>>(hR, F(42), yaC, 0);
  k_conv_v<<<NRESC, 96, 0, stream>>>(yvR, F(43), yvC);
  k_ac_out<<<NN, 64, 0, stream>>>(xa, xv, yaC, yvC, re, F(45), F(44), F(47), F(46), Ii(87));

  float* out_a = (float*)d_out;
  float* out_v = out_a + (size_t)NN * 64;
  k_copy<<<2048, 256, 0, stream>>>(out_a, xa, NN * 64);
  k_copy<<<3072, 256, 0, stream>>>(out_v, xv, NN * 96);

  k_msg_pre<<<NN, 64, 0, stream>>>(xa, xv, F(53), F(54), la, lv);
  k_msg<<<EE / 16, 32, 0, stream>>>(F(77), Ii(82), Ii(83), F(81), la, lv,
                                    hw, S_M0_Y000, F(56), F(58), F(60), out_a, out_v);
  k_msg_pre<<<NN, 64, 0, stream>>>(xa, xv, F(67), F(68), la, lv);
  k_msg<<<EE / 16, 32, 0, stream>>>(F(78), Ii(84), Ii(85), F(81), la, lv,
                                    hw, S_M1_Y000, F(70), F(72), F(74), out_a, out_v);
}
